// SpecificProfile_14302241096009
// MI455X (gfx1250) — matrix operand order repaired and checked by reading, not compiled, not run on hardware
//
#include <hip/hip_runtime.h>
#include <hip/hip_bf16.h>
#include <math.h>

typedef __attribute__((ext_vector_type(16))) _Float16     v16h;
typedef __attribute__((ext_vector_type(4)))  _Float16     v4h;
typedef __attribute__((ext_vector_type(8)))  float        v8f;
typedef __attribute__((ext_vector_type(4)))  float        v4f;
typedef __attribute__((ext_vector_type(8)))  unsigned int v8u;
typedef __attribute__((ext_vector_type(4)))  int          v4i;

#define KTOT    8
#define ALPHA   21
#define UCH     48
#define KW      168            // 8*21
#define KPAD    192            // pad K to 6 tiles of 32
#define WSTRIDE 200            // padded row stride (halves) -> LDS conflict-free
#define LSEQ    2000
#define PCOUNT  1993           // L - k + 1
#define NBATCH  576            // 8*12*6
#define XROW    (LSEQ*ALPHA)   // 42000 contiguous floats per batch
#define PBLK    512            // positions per block (8 waves * 4 tiles * 16 rows)
#define XCAP    10928          // >= 511*21 + 192, multiple of 4

#define S_ELEMS  (8*12*48)     // 4608
#define R_ELEMS  (8*21*48)     // 8064

#if __has_builtin(__builtin_amdgcn_global_load_async_to_lds_b128) && \
    __has_builtin(__builtin_amdgcn_s_wait_asynccnt)
#define HAVE_ASYNC_LDS 1
typedef __attribute__((address_space(1))) v4i* gv4p;   // global int4*
typedef __attribute__((address_space(3))) v4i* lv4p;   // LDS int4*
#endif

// float atomic max via int/uint trick; target initialized to -inf.
__device__ __forceinline__ void atomicMaxF(float* addr, float val) {
    if (val >= 0.0f) atomicMax((int*)addr, __float_as_int(val));
    else             atomicMin((unsigned int*)addr, __float_as_uint(val));
}

__global__ void sp_init_s(float* __restrict__ S) {
    int i = blockIdx.x * blockDim.x + threadIdx.x;
    if (i < S_ELEMS) S[i] = -INFINITY;
}

// Softmax over alphabet (21) per (j,u); write R (fp32) and transposed,
// K-padded f16 weights Wt[u][WSTRIDE] into workspace.
__global__ void sp_profiles(const float* __restrict__ Plogit,
                            const float* __restrict__ Q,
                            float* __restrict__ Rsec,
                            _Float16* __restrict__ Wt) {
    int idx = blockIdx.x * blockDim.x + threadIdx.x;   // 0..383
    if (idx >= KTOT * UCH) return;
    int u = idx % UCH;
    int j = idx / UCH;

    float vals[ALPHA];
    float mx = -INFINITY;
    #pragma unroll
    for (int a = 0; a < ALPHA; ++a) {
        vals[a] = Plogit[(j*ALPHA + a)*UCH + u];
        mx = fmaxf(mx, vals[a]);
    }
    float s = 0.0f;
    #pragma unroll
    for (int a = 0; a < ALPHA; ++a) { vals[a] = __expf(vals[a] - mx); s += vals[a]; }
    float inv = 1.0f / s;
    #pragma unroll
    for (int a = 0; a < ALPHA; ++a) {
        float P = vals[a] * inv;
        float r = __logf(fmaxf(P / Q[a], 1e-6f));
        Rsec[(j*ALPHA + a)*UCH + u] = r;
        Wt[u*WSTRIDE + (j*ALPHA + a)] = (_Float16)r;
    }
    if (j == 0) { // zero the K padding rows so WMMA K-tail contributes exactly 0
        for (int k = KW; k < WSTRIDE; ++k) Wt[u*WSTRIDE + k] = (_Float16)0.0f;
    }
}

// Main GEMM: Z[b,p,u] = sum_k X[b, p*21+k] * W[k,u]; masked max into S.
__global__ __launch_bounds__(256) void sp_conv_max(const float* __restrict__ X,
                                                   const _Float16* __restrict__ Wt,
                                                   float* __restrict__ Z,
                                                   float* __restrict__ S) {
    __shared__ __align__(16) _Float16 Wlds[UCH * WSTRIDE];   // 19200 B
    __shared__ __align__(16) _Float16 Xlds[XCAP];            // 21856 B
    __shared__ float smax[UCH];

    const int tid   = threadIdx.x;
    const int b     = blockIdx.y;              // 0..575
    const int pbase = blockIdx.x * PBLK;       // 0,512,1024,1536

    // Stage W into LDS: async DMA path (ASYNCcnt) when the toolchain has it.
#ifdef HAVE_ASYNC_LDS
    {
        // 19200 B = 1200 x 16B lane-chunks
        for (int i = tid; i < (UCH * WSTRIDE * 2) / 16; i += 256) {
            __builtin_amdgcn_global_load_async_to_lds_b128(
                (gv4p)((const char*)Wt + (size_t)i * 16),
                (lv4p)((char*)Wlds + (size_t)i * 16),
                0, 0);
        }
        if (tid < UCH) smax[tid] = -INFINITY;
    }
#else
    {
        const unsigned int* src = (const unsigned int*)Wt;
        unsigned int*       dst = (unsigned int*)Wlds;
        for (int i = tid; i < (UCH * WSTRIDE) / 2; i += 256) dst[i] = src[i];
        if (tid < UCH) smax[tid] = -INFINITY;
    }
#endif
    // Stage the contiguous X window, converting f32 -> f16 (vectorized x4).
    // base is 16B-aligned for all pbase; valid and XCAP are multiples of 4.
    {
        const float* Xb    = X + (size_t)b * XROW;
        const int    base  = pbase * ALPHA;
        const int    valid4 = (XROW - base) / 4;   // whole float4s valid
        const v4f*   Xb4   = (const v4f*)(Xb + base);
        v4h*         Xl4   = (v4h*)Xlds;
        for (int i = tid; i < XCAP / 4; i += 256) {
            v4f x = {};
            if (i < valid4) x = Xb4[i];
            Xl4[i] = __builtin_convertvector(x, v4h);
        }
    }
#ifdef HAVE_ASYNC_LDS
    __builtin_amdgcn_s_wait_asynccnt(0);
#endif
    __syncthreads();

    const int lane = tid & 31;
    const int w    = tid >> 5;
    const int m    = lane & 15;   // A-matrix row within tile
    const int hs   = lane >> 4;   // lane half-group
    const int n    = lane & 15;   // B/C column within N-tile
    float* Zb = Z + (size_t)b * (PCOUNT * UCH);

    // Per-lane running max for each of the 3 N-tiles (u = nt*16 + n).
    float mx3[3] = { -INFINITY, -INFINITY, -INFINITY };

    for (int s4 = 0; s4 < 4; ++s4) {
        const int tIdx = w * 4 + s4;          // 0..31 tiles in block
        const int pl0  = tIdx * 16;           // local position of tile row 0
        const int p0   = pbase + pl0;
        if (p0 >= PCOUNT) continue;           // wave-uniform: EXEC stays full

        const _Float16* xrow = &Xlds[(pl0 + m) * ALPHA];

        // A fragments for all 6 K-tiles (ISA 16-bit A 16x32 layout):
        // elements 0..7  <- K = kt*32 + hs*8 + e
        // elements 8..15 <- K = kt*32 + 16 + hs*8 + (e-8)
        v16h Af[6];
        #pragma unroll
        for (int kt = 0; kt < 6; ++kt) {
            const int kb = kt * 32 + hs * 8;
            #pragma unroll
            for (int e = 0; e < 8; ++e) {
                Af[kt][e]     = xrow[kb + e];
                Af[kt][e + 8] = xrow[kb + 16 + e];
            }
        }

        #pragma unroll
        for (int nt = 0; nt < 3; ++nt) {
            v8f acc = {};
            // B fragment (Kx16 f16): elements 0..7 hold K=kb+hs*8+e of column n,
            // elements 8..15 hold K=kb+16+hs*8+(e-8); 2 halves per VGPR.
            const unsigned int* wrow =
                (const unsigned int*)&Wlds[(nt * 16 + n) * WSTRIDE];
            #pragma unroll
            for (int kt = 0; kt < 6; ++kt) {
                union { v8u u; v16h h; } bf;
                const int kb2 = kt * 16 + hs * 4;   // u32 index
                #pragma unroll
                for (int i = 0; i < 8; ++i) bf.u[i] = wrow[kb2 + (i & 3) + (i >> 2) * 8];
                acc = __builtin_amdgcn_wmma_f32_16x16x32_f16(
                          false, Af[kt], false, bf.h, (short)0, acc, false, false);
            }
            // C/D layout: VGPR v holds row M = v + 8*hs, column N = n.
            float  mx   = -INFINITY;
            float* zcol = &Zb[(size_t)p0 * UCH + nt * 16 + n];
            if (p0 + 16 <= PCOUNT) {
                // Interior tile: all 16 rows valid -> no per-row predication.
                #pragma unroll
                for (int v = 0; v < 8; ++v) {
                    __builtin_nontemporal_store(acc[v],
                        zcol + (size_t)(v + 8 * hs) * UCH);
                    mx = fmaxf(mx, acc[v]);
                }
            } else {
                #pragma unroll
                for (int v = 0; v < 8; ++v) {
                    if (p0 + v + 8 * hs < PCOUNT) {
                        __builtin_nontemporal_store(acc[v],
                            zcol + (size_t)(v + 8 * hs) * UCH);
                        mx = fmaxf(mx, acc[v]);
                    }
                }
            }
            mx3[nt] = fmaxf(mx3[nt], mx);
        }
    }
    // One LDS float-atomic-max per N-tile per lane (instead of per tile).
    #pragma unroll
    for (int nt = 0; nt < 3; ++nt) atomicMaxF(&smax[nt * 16 + n], mx3[nt]);

    __syncthreads();
    if (tid < UCH) atomicMaxF(&S[(b / 6) * UCH + tid], smax[tid]);
}

extern "C" void kernel_launch(void* const* d_in, const int* in_sizes, int n_in,
                              void* d_out, int out_size, void* d_ws, size_t ws_size,
                              hipStream_t stream) {
    const float* X      = (const float*)d_in[0];   // [8,12,6,2000,21]
    const float* Plogit = (const float*)d_in[1];   // [8,21,48]
    const float* Q      = (const float*)d_in[2];   // [21]

    float* out = (float*)d_out;
    float* S   = out;                       // 4608
    float* R   = out + S_ELEMS;             // 8064
    float* Z   = out + S_ELEMS + R_ELEMS;   // 576*1993*48

    _Float16* Wt = (_Float16*)d_ws;         // [48][200] f16, 19200 B

    sp_init_s  <<<(S_ELEMS + 255) / 256, 256, 0, stream>>>(S);
    sp_profiles<<<3, 128, 0, stream>>>(Plogit, Q, R, Wt);
    sp_conv_max<<<dim3(4, NBATCH), 256, 0, stream>>>(X, Wt, Z, S);
}
